// LEAF_90374701843168
// MI455X (gfx1250) — matrix operand order repaired and checked by reading, not compiled, not run on hardware
//
#include <hip/hip_runtime.h>
#include <hip/hip_bf16.h>
#include <math.h>

// ---------------------------------------------------------------------------
// LEAF frontend for MI455X (gfx1250, wave32, WMMA).
//   Stage 1: build Gabor weights directly in WMMA-A-fragment order (bf16)
//   Stage 2: Gabor conv as implicit GEMM with v_wmma_f32_16x16x32_bf16,
//            2 N-tiles per wave (A reuse), fused |.|^2 power epilogue
//   Stage 3: EMA linear-recurrence scan + PCEN, in-place
//   Stage 4: depthwise Gaussian pool (stride 160) + bias
//   Stage 5: instance norm over time
// ---------------------------------------------------------------------------

typedef __attribute__((ext_vector_type(16))) __bf16 v16bf;
typedef __attribute__((ext_vector_type(8)))  float  v8f;

#define F_FILT 40
#define NROWS  80          // 2*F_FILT (cos/sin interleaved)
#define KG     401
#define KPOOL  401
#define HOP    160
#define TT     160000
#define NB     8
#define MTILES 5           // 80 / 16
#define KCH    13          // ceil(401/32)
#define NTAU   1000

// workspace layout (bytes)
#define OFF_WFRAG 0u              // 5*13*32*16 bf16 = 66560 B
#define OFF_POOL  66560u          // 40*401 f32     = 64160 B
#define OFF_POWER 131072u         // 8*40*160000 f32 = 204.8 MB

// ---------------------------------------------------------------------------
// Stage 1a: Gabor weights, written directly in per-lane WMMA A-fragment order.
// A (16-bit, 16x32): lane L -> M = L&15, half = L>>4;
// element j: K = 16*half + j (contiguous k order chosen here; B uses the same)
// flat layout: wfrag[((m*13+kc)*32 + lane)*16 + j]
// ---------------------------------------------------------------------------
__global__ __launch_bounds__(256)
void build_wfrag_kernel(const float* __restrict__ eta,
                        const float* __restrict__ sigma,
                        __bf16* __restrict__ wfrag)
{
    int idx = blockIdx.x * 256 + threadIdx.x;
    if (idx >= MTILES * KCH * 32 * 16) return;
    int j    = idx & 15;
    int lane = (idx >> 4) & 31;
    int rest = idx >> 9;            // m*13 + kc
    int kc   = rest % KCH;
    int m    = rest / KCH;
    int row  = m * 16 + (lane & 15);
    int half = lane >> 4;
    int v    = j >> 1, p = j & 1;
    int K    = 16 * half + j;
    int k    = kc * 32 + K;
    float w = 0.0f;
    if (k < KG) {
        int f = row >> 1;
        float tk  = (float)k - 200.0f;                    // (K_GABOR-1)/2
        float sg  = sigma[f];
        float u   = tk / sg;
        float env = __expf(-0.5f * u * u) / (2.50662827463100050242f * sg);
        float ph  = eta[f] * tk;
        w = env * (((row & 1) == 0) ? __cosf(ph) : __sinf(ph));
    }
    wfrag[idx] = (__bf16)w;
}

// Stage 1b: Gaussian pooling weights (f32, exact)
__global__ __launch_bounds__(256)
void build_pool_kernel(const float* __restrict__ bw, float* __restrict__ pk)
{
    int idx = blockIdx.x * 256 + threadIdx.x;
    if (idx >= F_FILT * KPOOL) return;
    int f = idx / KPOOL, k = idx % KPOOL;
    float b   = fminf(fmaxf(bw[f], 2.0f / (float)KPOOL), 0.5f);
    float den = b * 0.5f * (float)(KPOOL - 1);
    float tp  = (float)k - 201.0f;                        // arange(-201, 200)
    float u   = tp / den;
    pk[idx] = __expf(-0.5f * u * u);
}

// ---------------------------------------------------------------------------
// Stage 2: implicit GEMM, Y[80,T] = W[80,401] x Xwin[401,T], via WMMA bf16.
// Block = 1 batch x 256 time steps; wave = 2 N-tiles of 16 cols; 13 K-chunks.
// B fragment, same contiguous k order as A (lanes 0-15: K=0..15, lanes 16-31: K=16..31, col = L&15)
// for (wave w, tile u) is 16 *contiguous* bf16 at xs[kc*32 + 32w + 16u + L].
// Per K-chunk: 5 A-fragments batched into registers, then 10 back-to-back
// WMMAs (A reused across the two N-tiles) -> one loadcnt wait per chunk.
// Epilogue: power[f] = Y[2f]^2 + Y[2f+1]^2 (adjacent C VGPRs).
// ---------------------------------------------------------------------------
__global__ __launch_bounds__(256)
void gabor_power_kernel(const float* __restrict__ x,
                        const __bf16* __restrict__ wfrag,
                        float* __restrict__ powerbuf)
{
    __shared__ __bf16 xs[672];   // covers t in [tblk-400, tblk+271]

    const int b    = blockIdx.x / 625;
    const int tblk = (blockIdx.x % 625) * 256;
    const int tid  = threadIdx.x;

    for (int i = tid; i < 672; i += 256) {
        int t = tblk - 400 + i;
        float v = (t >= 0 && t < TT) ? x[(size_t)b * TT + t] : 0.0f;
        xs[i] = (__bf16)v;
    }
    __syncthreads();

    const int wave = tid >> 5;     // 8 wave32 waves
    const int lane = tid & 31;
    const int boff = wave * 32 + lane;

    v8f acc0[MTILES], acc1[MTILES];
    v8f zero = {};
    #pragma unroll
    for (int m = 0; m < MTILES; ++m) { acc0[m] = zero; acc1[m] = zero; }

    #pragma unroll
    for (int kc = 0; kc < KCH; ++kc) {
        // two B fragments (N-tile 0 and 1), each 16 contiguous bf16 in LDS
        v16bf bf0, bf1;
        const int base = kc * 32 + boff;
        #pragma unroll
        for (int j = 0; j < 16; ++j) {
            bf0[j] = xs[base + j];
            bf1[j] = xs[base + 16 + j];
        }

        // batch all 5 A fragments into distinct registers (one wait, clauses)
        v16bf af[MTILES];
        #pragma unroll
        for (int m = 0; m < MTILES; ++m)
            af[m] = *(const v16bf*)(wfrag + ((size_t)(m * KCH + kc) * 32 + lane) * 16);

        // 10 WMMAs back-to-back; A reused across the two N-tiles
        #pragma unroll
        for (int m = 0; m < MTILES; ++m)
            acc0[m] = __builtin_amdgcn_wmma_f32_16x16x32_bf16(
                false, af[m], false, bf0, (short)0, acc0[m], false, false);
        #pragma unroll
        for (int m = 0; m < MTILES; ++m)
            acc1[m] = __builtin_amdgcn_wmma_f32_16x16x32_bf16(
                false, af[m], false, bf1, (short)0, acc1[m], false, false);
    }

    // C layout: VGPR r -> M = r + 8*half, col N = lane&15.
    const int n    = lane & 15;
    const int half = lane >> 4;
    const int t0   = tblk + wave * 32 + n;
    float* prow = powerbuf + (size_t)b * F_FILT * TT + t0;
    #pragma unroll
    for (int m = 0; m < MTILES; ++m) {
        #pragma unroll
        for (int r = 0; r < 8; r += 2) {
            int f = m * 8 + (r >> 1) + 4 * half;
            float re0 = acc0[m][r], im0 = acc0[m][r + 1];
            float re1 = acc1[m][r], im1 = acc1[m][r + 1];
            prow[(size_t)f * TT]      = re0 * re0 + im0 * im0;
            prow[(size_t)f * TT + 16] = re1 * re1 + im1 * im1;
        }
    }
}

// ---------------------------------------------------------------------------
// Stage 3: EMA scan (y_t = (1-s) y_{t-1} + s p_t, y_0 = p_0) + PCEN, in-place.
// One workgroup per (b,f) row: 256 chunks of 625, two-pass chunked scan.
// ---------------------------------------------------------------------------
__global__ __launch_bounds__(256)
void ema_pcen_kernel(float* __restrict__ buf,
                     const float* __restrict__ ema_w,
                     const float* __restrict__ alpha,
                     const float* __restrict__ delta,
                     const float* __restrict__ root)
{
    const int row = blockIdx.x;          // b*40 + f
    const int f   = row % F_FILT;
    const int tid = threadIdx.x;
    float* p = buf + (size_t)row * TT;

    const float s    = fminf(fmaxf(ema_w[f], 0.0f), 1.0f);
    const float a    = fminf(alpha[f], 1.0f);
    const float invr = 1.0f / fmaxf(root[f], 1.0f);
    const float d    = delta[f];
    const float droot = __powf(d, invr);

    __shared__ float As[256], Bs[256], inc[256];

    const int c0 = tid * 625, c1 = c0 + 625;

    // pass 1: per-chunk affine transform (A, B): y_out = A*y_in + B
    float A = 1.0f, Bc = 0.0f;
    for (int t = c0; t < c1; ++t) {
        float xv = p[t];
        float at = (t == 0) ? 0.0f : (1.0f - s);
        float bt = (t == 0) ? xv   : s * xv;
        A  = at * A;
        Bc = at * Bc + bt;
    }
    As[tid] = A; Bs[tid] = Bc;
    __syncthreads();

    if (tid == 0) {
        float y = 0.0f;
        for (int i = 0; i < 256; ++i) { inc[i] = y; y = As[i] * y + Bs[i]; }
    }
    __syncthreads();

    // pass 2: replay recurrence with incoming state, fuse PCEN, write in-place
    float y = inc[tid];
    for (int t = c0; t < c1; ++t) {
        float xv = p[t];
        float at = (t == 0) ? 0.0f : (1.0f - s);
        float bt = (t == 0) ? xv   : s * xv;
        y = at * y + bt;
        float comp = __powf(xv / __powf(1e-6f + y, a) + d, invr) - droot;
        p[t] = comp;
    }
}

// ---------------------------------------------------------------------------
// Stage 4: depthwise Gaussian pool, stride 160, left-pad 400, + bias
// ---------------------------------------------------------------------------
__global__ __launch_bounds__(256)
void pool_kernel(const float* __restrict__ comp,
                 const float* __restrict__ pk,
                 const float* __restrict__ bias,
                 float* __restrict__ out)
{
    const int row = blockIdx.x;          // b*40 + f
    const int f   = row % F_FILT;
    __shared__ float pks[KPOOL];
    for (int i = threadIdx.x; i < KPOOL; i += 256) pks[i] = pk[f * KPOOL + i];
    __syncthreads();

    const float* c  = comp + (size_t)row * TT;
    const float  bi = bias[f];
    for (int tau = threadIdx.x; tau < NTAU; tau += 256) {
        float acc = 0.0f;
        const int start = tau * HOP - 400;
        const int k0 = (start < 0) ? -start : 0;   // left zero-pad
        for (int k = k0; k < KPOOL; ++k) {
            acc += pks[k] * c[start + k];
        }
        out[(size_t)row * NTAU + tau] = acc + bi;
    }
}

// ---------------------------------------------------------------------------
// Stage 5: instance norm over time (mean/var over 1000), in-place on d_out
// ---------------------------------------------------------------------------
__global__ __launch_bounds__(256)
void norm_kernel(float* __restrict__ out)
{
    const int row = blockIdx.x;
    float* o = out + (size_t)row * NTAU;
    __shared__ float ssum[256], ssq[256];
    float s = 0.0f, q = 0.0f;
    for (int i = threadIdx.x; i < NTAU; i += 256) {
        float v = o[i]; s += v; q += v * v;
    }
    ssum[threadIdx.x] = s; ssq[threadIdx.x] = q;
    __syncthreads();
    for (int off = 128; off > 0; off >>= 1) {
        if (threadIdx.x < off) {
            ssum[threadIdx.x] += ssum[threadIdx.x + off];
            ssq[threadIdx.x]  += ssq[threadIdx.x + off];
        }
        __syncthreads();
    }
    const float mean = ssum[0] * (1.0f / NTAU);
    const float var  = ssq[0] * (1.0f / NTAU) - mean * mean;
    const float inv  = rsqrtf(var + 1e-5f);
    for (int i = threadIdx.x; i < NTAU; i += 256) {
        o[i] = (o[i] - mean) * inv;
    }
}

// ---------------------------------------------------------------------------
extern "C" void kernel_launch(void* const* d_in, const int* in_sizes, int n_in,
                              void* d_out, int out_size, void* d_ws, size_t ws_size,
                              hipStream_t stream)
{
    (void)in_sizes; (void)n_in; (void)out_size; (void)ws_size;
    const float* x     = (const float*)d_in[0];
    const float* eta   = (const float*)d_in[1];
    const float* sigma = (const float*)d_in[2];
    const float* bw    = (const float*)d_in[3];
    const float* bias  = (const float*)d_in[4];
    const float* emaw  = (const float*)d_in[5];
    const float* alpha = (const float*)d_in[6];
    const float* delta = (const float*)d_in[7];
    const float* root  = (const float*)d_in[8];
    float* out = (float*)d_out;

    char* ws = (char*)d_ws;
    __bf16* wfrag    = (__bf16*)(ws + OFF_WFRAG);
    float*  pk       = (float*)(ws + OFF_POOL);
    float*  powerbuf = (float*)(ws + OFF_POWER);

    build_wfrag_kernel<<<(MTILES * KCH * 32 * 16 + 255) / 256, 256, 0, stream>>>(
        eta, sigma, wfrag);
    build_pool_kernel<<<(F_FILT * KPOOL + 255) / 256, 256, 0, stream>>>(bw, pk);

    gabor_power_kernel<<<NB * 625, 256, 0, stream>>>(x, wfrag, powerbuf);

    ema_pcen_kernel<<<NB * F_FILT, 256, 0, stream>>>(powerbuf, emaw, alpha, delta, root);
    pool_kernel<<<NB * F_FILT, 256, 0, stream>>>(powerbuf, pk, bias, out);
    norm_kernel<<<NB * F_FILT, 256, 0, stream>>>(out);
}
